// WordAttn_21818433863898
// MI455X (gfx1250) — matrix operand order repaired and checked by reading, not compiled, not run on hardware
//
#include <hip/hip_runtime.h>
#include <hip/hip_bf16.h>

#define BB 256
#define TT 512
#define DD 256

typedef __attribute__((ext_vector_type(16))) __bf16 v16bf;
typedef __attribute__((ext_vector_type(2)))  __bf16 v2bf;
typedef __attribute__((ext_vector_type(16))) float  v16f;
typedef __attribute__((ext_vector_type(8)))  float  v8f;
typedef __attribute__((ext_vector_type(2)))  float  v2f;

// fp32 pair -> packed bf16 dword via native packed convert (RNE)
__device__ __forceinline__ unsigned pk_bf16(float a, float b) {
    v2f  f = {a, b};
    v2bf r = __builtin_convertvector(f, v2bf);
    return __builtin_bit_cast(unsigned, r);
}

// hardware tanh if the builtin exists; otherwise branchless trans-op form.
// Device-only builtins are hidden from the host compilation pass.
__device__ __forceinline__ float fast_tanh(float x) {
#if defined(__HIP_DEVICE_COMPILE__)
# if __has_builtin(__builtin_amdgcn_tanhf)
    return __builtin_amdgcn_tanhf(x);
# elif __has_builtin(__builtin_amdgcn_tanh_f32)
    return __builtin_amdgcn_tanh_f32(x);
# else
    // tanh(x) = 1 - 2/(e^{2x}+1); v_exp_f32 computes 2^x -> scale by 2*log2(e).
    // Large +x: exp2 -> inf -> rcp -> 0 -> 1.  Large -x: exp2 -> 0 -> -1.
    float ex = __builtin_amdgcn_exp2f(x * 2.885390081777927f);
    return 1.0f - 2.0f * __builtin_amdgcn_rcpf(ex + 1.0f);
# endif
#else
    return tanhf(x);   // host pass only; never runs on device
#endif
}

extern "C" __global__ __launch_bounds__(256)
void word_attn_kernel(const float* __restrict__ ip,
                      const float* __restrict__ W,
                      const float* __restrict__ bias,
                      const float* __restrict__ ctx,
                      float* __restrict__ out)
{
    extern __shared__ char smem[];
    // W in bf16, pre-swizzled to WMMA B-fragment layout:
    // [kt(8)][et(16)][lane(32)][vgpr(8)] dwords = 128 KB
    unsigned* WB    = (unsigned*)smem;
    float*    biasL = (float*)(smem + 131072);   // 256 f32
    float*    ctxL  = biasL + 256;               // 256 f32
    float*    sc    = ctxL  + 256;               // 512 scores -> attn
    float*    red   = sc    + 512;               // 256 reduction scratch

    const int tid  = threadIdx.x;
    const int b    = blockIdx.x;
    const int lane = tid & 31;
    const int wave = tid >> 5;
    const int hlf  = lane >> 4;    // lane half (K interleave select)
    const int mrow = lane & 15;

    // ---- Prologue: W[e][k] -> LDS bf16 B-fragments ----
    // B[k][e] = W[e][k]; fragment layout: lane n<16 -> N=n, K=0..15 ;
    // lane n>=16 -> N=n-16, K=16..31 ; VGPR v holds K=2v (lo), 2v+1 (hi).
    for (int i = 0; i < 128; ++i) {
        int p = i * 256 + tid;               // coalesced over tid
        int e = p >> 7;
        int k = (p & 127) << 1;
        float2 wv = *(const float2*)(W + e * DD + k);
        int kt = k >> 5, kk = k & 31;
        int n  = ((kk >= 16) ? 16 : 0) + (e & 15);
        int v  = (kk & 15) >> 1;
        int et = e >> 4;
        WB[((kt * 16 + et) * 32 + n) * 8 + v] = pk_bf16(wv.x, wv.y);
    }
    biasL[tid] = bias[tid];
    ctxL[tid]  = ctx[tid];
    __syncthreads();

    // ---- Phase 1: scores[t] = sum_e tanh(ip@W^T + b)[t,e] * ctx[e] ----
    const float* ipb = ip + (size_t)b * TT * DD;

    for (int mt = wave; mt < TT / 16; mt += 8) {
        const int t0 = mt * 16;
        const float* rowp = ipb + (size_t)(t0 + mrow) * DD;

        // A-fragments for full K=256 (8 k-tiles), converted fp32->bf16 once.
        // A layout used here (contiguous K order, matching WB): lanes 0-15 hold K off+0..15
        // with off = 0; lanes 16-31 with off = 16.
        v16bf a[8];
#pragma unroll
        for (int kt = 0; kt < 8; ++kt) {
            int k0 = kt * 32 + hlf * 16;
            v8f lo = *(const v8f*)(rowp + k0);        // K off..off+7
            v8f hi = *(const v8f*)(rowp + k0 + 8);    // K off+8..off+15
            v16f f = __builtin_shufflevector(lo, hi,
                        0,1,2,3,4,5,6,7,8,9,10,11,12,13,14,15);
            a[kt] = __builtin_convertvector(f, v16bf); // packed bf16 cvt x8
        }

        v8f sacc = {0.f, 0.f, 0.f, 0.f, 0.f, 0.f, 0.f, 0.f};
        for (int et = 0; et < 16; ++et) {
            v8f c = {};
#pragma unroll
            for (int kt = 0; kt < 8; ++kt) {
                const v16bf bf =
                    *(const v16bf*)(&WB[((kt * 16 + et) * 32 + lane) * 8]);
                c = __builtin_amdgcn_wmma_f32_16x16x32_bf16(
                        false, a[kt], false, bf, (short)0, c, false, false);
            }
            const int   e  = et * 16 + mrow;
            const float be = biasL[e];
            const float ce = ctxL[e];
#pragma unroll
            for (int r = 0; r < 8; ++r) {
                float h = fast_tanh(c[r] + be);
                sacc[r] += h * ce;
            }
        }
        // reduce over the 16 lanes of each half (rows differ per half)
#pragma unroll
        for (int r = 0; r < 8; ++r) {
            float s = sacc[r];
            s += __shfl_xor(s, 1);
            s += __shfl_xor(s, 2);
            s += __shfl_xor(s, 4);
            s += __shfl_xor(s, 8);
            sacc[r] = s;
        }
        if (mrow == 0) {
#pragma unroll
            for (int r = 0; r < 8; ++r)
                sc[t0 + hlf * 8 + r] = sacc[r];   // M = r (half0) / 8+r (half1)
        }
    }
    __syncthreads();

    // ---- Phase 2: softmax over T=512 ----
    float s0 = sc[tid], s1 = sc[tid + 256];
    red[tid] = fmaxf(s0, s1);
    __syncthreads();
    for (int s = 128; s > 0; s >>= 1) {
        if (tid < s) red[tid] = fmaxf(red[tid], red[tid + s]);
        __syncthreads();
    }
    float gmax = red[0];
    __syncthreads();
    float e0 = __expf(s0 - gmax), e1 = __expf(s1 - gmax);
    red[tid] = e0 + e1;
    __syncthreads();
    for (int s = 128; s > 0; s >>= 1) {
        if (tid < s) red[tid] += red[tid + s];
        __syncthreads();
    }
    float inv = 1.0f / red[0];
    __syncthreads();
    sc[tid]       = e0 * inv;
    sc[tid + 256] = e1 * inv;
    __syncthreads();

    // ---- Phase 3: out[b,d] = sum_t attn[t] * ip[b,t,d]  (fp32, L2-resident) ----
    float acc = 0.f;
#pragma unroll 8
    for (int t = 0; t < TT; ++t)
        acc = fmaf(sc[t], ipb[(size_t)t * DD + tid], acc);
    out[b * DD + tid] = acc;
}

extern "C" void kernel_launch(void* const* d_in, const int* in_sizes, int n_in,
                              void* d_out, int out_size, void* d_ws, size_t ws_size,
                              hipStream_t stream) {
    const float* ip   = (const float*)d_in[0];
    const float* W    = (const float*)d_in[1];
    const float* bias = (const float*)d_in[2];
    const float* ctx  = (const float*)d_in[3];
    float* out = (float*)d_out;

    const size_t shmem = 131072 + (256 + 256 + 512 + 256) * sizeof(float);
    // Opt in to >64KB dynamic LDS (WGP has 320KB); harmless if already set.
    (void)hipFuncSetAttribute((const void*)word_attn_kernel,
                              hipFuncAttributeMaxDynamicSharedMemorySize,
                              (int)shmem);
    word_attn_kernel<<<BB, 256, shmem, stream>>>(ip, W, bias, ctx, out);
}
